// SLIM_57123065037314
// MI455X (gfx1250) — matrix operand order repaired and checked by reading, not compiled, not run on hardware
//
#include <hip/hip_runtime.h>
#include <hip/hip_bf16.h>
#include <math.h>

typedef __attribute__((ext_vector_type(8)))  float          v8f;
typedef __attribute__((ext_vector_type(16))) __bf16         v16bf;
typedef __attribute__((ext_vector_type(16))) unsigned short v16u;

#define FSTR 128   // padded stride of the [N, 97] hop-concat buffer

static __device__ __forceinline__ unsigned short f2bf(float f) {
    union { float f; unsigned int u; } cv; cv.f = f;
    unsigned int u = cv.u;
    unsigned int r = u + 0x7FFFu + ((u >> 16) & 1u);   // round-to-nearest-even
    return (unsigned short)(r >> 16);
}

// ---------------------------------------------------------------- init deg/e2n
__global__ void init_deg_e2n(float* __restrict__ deg, float* __restrict__ e2n, int N) {
    int i = blockIdx.x * blockDim.x + threadIdx.x;   // over N*32
    if (i < N * 32) {
        e2n[i] = 0.f;
        if ((i & 31) == 0) deg[i >> 5] = 1.0f;       // self-loop degree
    }
}

// ------------------------------------------------------- edge_feat -> e2n, deg
__global__ void edge_e2n(const float* __restrict__ ef, const int* __restrict__ dst,
                         float* __restrict__ e2n, float* __restrict__ deg, int E) {
    int i = blockIdx.x * blockDim.x + threadIdx.x;   // over E*32
    if (i >= E * 32) return;
    int e = i >> 5, c = i & 31;
    int d = dst[e];
    atomicAdd(&e2n[(size_t)d * 32 + c], ef[(size_t)e * 32 + c]);
    if (c == 0) atomicAdd(&deg[d], 1.0f);
}

// ------------------------------------------- layer-0 pooled init (virtual concat)
__global__ void pooled_init0(const float* __restrict__ nf, const float* __restrict__ e2n,
                             float* __restrict__ pooled, int N) {
    long i = (long)blockIdx.x * blockDim.x + threadIdx.x;   // over N*160
    if (i >= (long)N * 160) return;
    int n = (int)(i / 160), c = (int)(i % 160);
    pooled[i] = (c < 128) ? nf[(size_t)n * 128 + c] : e2n[(size_t)n * 32 + (c - 128)];
}

// ---------------------------------------------------- layer-0 edge scatter (160)
__global__ void scatter0(const float* __restrict__ nf, const float* __restrict__ e2n,
                         const int* __restrict__ src, const int* __restrict__ dst,
                         float* __restrict__ pooled, int E) {
    long i = (long)blockIdx.x * blockDim.x + threadIdx.x;   // over E*160
    if (i >= (long)E * 160) return;
    int e = (int)(i / 160), c = (int)(i % 160);
    int s = src[e], d = dst[e];
    float v = (c < 128) ? nf[(size_t)s * 128 + c] : e2n[(size_t)s * 32 + (c - 128)];
    atomicAdd(&pooled[(size_t)d * 160 + c], v);
}

// ---------------------------------------------- later layers pooled init (32 col)
__global__ void pooled_initL(const float* __restrict__ feat, float* __restrict__ pooled,
                             int N, int off) {
    int i = blockIdx.x * blockDim.x + threadIdx.x;   // over N*32
    if (i >= N * 32) return;
    int n = i >> 5, c = i & 31;
    pooled[i] = feat[(size_t)n * FSTR + off + c];
}

// --------------------------------------------------- later layers edge scatter
__global__ void scatterL(const float* __restrict__ feat, const int* __restrict__ src,
                         const int* __restrict__ dst, float* __restrict__ pooled,
                         int E, int off) {
    int i = blockIdx.x * blockDim.x + threadIdx.x;   // over E*32
    if (i >= E * 32) return;
    int e = i >> 5, c = i & 31;
    atomicAdd(&pooled[(size_t)dst[e] * 32 + c], feat[(size_t)src[e] * FSTR + off + c]);
}

// --------------------------------------------------------------- WMMA GEMM layer
// out[n, j] = tanh( (sum_k pooled[n,k] * W[k,j] + b[j]) / deg[n] )  -> feat cols
// One wave per 16-node tile; 8 waves / block -> 128 nodes per block.
// A-tile (16x32 f32, 2KB) is filled with GLOBAL_LOAD_ASYNC_TO_LDS_B128 (ASYNCcnt
// path, no VGPR round-trip); each wave waits its own ASYNCcnt, tile is
// wave-private so no block barrier is required.
template <int DIN, int DOUT>
__global__ __launch_bounds__(256)
void gnn_gemm_kernel(const float* __restrict__ pooled, const float* __restrict__ W,
                     const float* __restrict__ bias, const float* __restrict__ deg,
                     float* __restrict__ feat, int col_off) {
    constexpr int KSTEPS = (DIN + 31) / 32;
    constexpr int NTILES = (DOUT + 15) / 16;
    __shared__ __attribute__((aligned(16))) float atile[8][512];  // per-wave 16x32 f32

    const int lane    = threadIdx.x & 31;
    const int wave    = threadIdx.x >> 5;
    const int rowBase = blockIdx.x * 128 + wave * 16;
    const int m = lane & 15;                        // row within A-frag / col within B-frag
    const int g = lane >> 4;                        // lane group

    // async-copy geometry: 4 x b128 issues/wave move the 2KB tile
    const int arow = lane >> 3;                     // + 4*j
    const int acol = (lane & 7) * 4;                // float column within 32-wide chunk
    const unsigned ldsbase = (unsigned)(size_t)(&atile[wave][0]);

    const v8f vzero = {0.f, 0.f, 0.f, 0.f, 0.f, 0.f, 0.f, 0.f};
    v8f acc[NTILES];
#pragma unroll
    for (int t = 0; t < NTILES; ++t) acc[t] = vzero;

    for (int s = 0; s < KSTEPS; ++s) {
        // ---- async fill of the wave-private A tile (memory -> LDS direct) ----
#pragma unroll
        for (int j = 0; j < 4; ++j) {
            const int row = j * 4 + arow;
            unsigned long long ga = (unsigned long long)(size_t)
                (pooled + (size_t)(rowBase + row) * DIN + s * 32 + acol);
            unsigned la = ldsbase + (unsigned)((row * 32 + acol) * 4);
            asm volatile("global_load_async_to_lds_b128 %0, %1, off"
                         :: "v"(la), "v"(ga) : "memory");
        }
        asm volatile("s_wait_asynccnt 0x0" ::: "memory");

        // A fragment: 16-bit A 16x32 layout (ISA 7.12.2)
        v16u au;
#pragma unroll
        for (int e = 0; e < 16; ++e) {
            int v = e >> 1, h = e & 1;
            int k = (v < 4) ? (g * 8 + v * 2 + h) : (16 + g * 8 + (v - 4) * 2 + h);
            au[e] = f2bf(atile[wave][m * 32 + k]);
        }
        v16bf afrag = __builtin_bit_cast(v16bf, au);

#pragma unroll
        for (int t = 0; t < NTILES; ++t) {
            // B fragment: 32x16, k = laneGroup*8 + elem (elem<8), 16 + laneGroup*8 + elem-8 (elem>=8), n = lane&15
            v16u bu;
            const int n = t * 16 + m;
#pragma unroll
            for (int e = 0; e < 16; ++e) {
                int k = s * 32 + ((e < 8) ? (g * 8 + e) : (16 + g * 8 + (e - 8)));
                float w = (k < DIN && n < DOUT) ? W[k * DOUT + n] : 0.f;
                bu[e] = f2bf(w);
            }
            v16bf bfrag = __builtin_bit_cast(v16bf, bu);
            acc[t] = __builtin_amdgcn_wmma_f32_16x16x32_bf16(
                false, afrag, false, bfrag, (short)0, acc[t], false, false);
        }
    }

    // Epilogue: C/D layout -> row = r + g*8, col = lane&15
#pragma unroll
    for (int t = 0; t < NTILES; ++t) {
        const int n  = t * 16 + m;
        const float bv = (n < DOUT) ? bias[n] : 0.f;
#pragma unroll
        for (int r = 0; r < 8; ++r) {
            const int node = rowBase + g * 8 + r;
            const float y = tanhf((acc[t][r] + bv) / deg[node]);
            if (n < DOUT) feat[(size_t)node * FSTR + col_off + n] = y;
        }
    }
}

// ------------------------------------------------ SortPool + conv tail, 1 blk/graph
__global__ __launch_bounds__(128)
void tail_kernel(const float* __restrict__ feat,
                 const float* __restrict__ wc1, const float* __restrict__ bc1,
                 const float* __restrict__ wc2, const float* __restrict__ bc2,
                 const float* __restrict__ wo,  const float* __restrict__ bo,
                 float* __restrict__ out) {
    __shared__ float keys[1024];
    __shared__ int   sel[30];
    __shared__ float sp[30][97];
    __shared__ float x1[480];     // [16][30]
    __shared__ float xp[240];     // [16][15]
    __shared__ float x2[352];     // [32][11]
    __shared__ float rv[128];
    __shared__ int   ri[128];

    const int g = blockIdx.x, tid = threadIdx.x;
    const size_t nb = (size_t)g * 1024;

    for (int i = tid; i < 1024; i += 128) keys[i] = feat[(nb + i) * FSTR + 96];
    __syncthreads();

    // top-30 by iterative argmax (stable: ties -> lowest index, matches lax.top_k)
    for (int j = 0; j < 30; ++j) {
        float bv = -INFINITY; int bi = 1 << 30;
        for (int i = tid; i < 1024; i += 128) {
            float v = keys[i];
            if (v > bv || (v == bv && i < bi)) { bv = v; bi = i; }
        }
        rv[tid] = bv; ri[tid] = bi;
        __syncthreads();
        for (int sft = 64; sft > 0; sft >>= 1) {
            if (tid < sft) {
                float v = rv[tid + sft]; int i2 = ri[tid + sft];
                if (v > rv[tid] || (v == rv[tid] && i2 < ri[tid])) { rv[tid] = v; ri[tid] = i2; }
            }
            __syncthreads();
        }
        if (tid == 0) { sel[j] = ri[0]; keys[ri[0]] = -INFINITY; }
        __syncthreads();
    }

    // gather top-30 rows (97 hop features)
    for (int t = tid; t < 30 * 97; t += 128) {
        int j = t / 97, c = t % 97;
        sp[j][c] = feat[(nb + sel[j]) * FSTR + c];
    }
    __syncthreads();

    // conv1: stride-D conv == per-position 97->16 linear, relu
    for (int t = tid; t < 480; t += 128) {
        int c = t / 30, p = t % 30;
        float s = bc1[c];
        for (int d = 0; d < 97; ++d) s += wc1[c * 97 + d] * sp[p][d];
        x1[c * 30 + p] = fmaxf(s, 0.f);
    }
    __syncthreads();

    for (int t = tid; t < 240; t += 128) {           // maxpool 2,2
        int c = t / 15, q = t % 15;
        xp[c * 15 + q] = fmaxf(x1[c * 30 + 2 * q], x1[c * 30 + 2 * q + 1]);
    }
    __syncthreads();

    for (int t = tid; t < 352; t += 128) {           // conv2 k=5, relu
        int c2 = t / 11, p = t % 11;
        float s = bc2[c2];
        for (int c1 = 0; c1 < 16; ++c1)
            for (int k = 0; k < 5; ++k)
                s += wc2[c2 * 80 + c1 * 5 + k] * xp[c1 * 15 + p + k];
        x2[t] = fmaxf(s, 0.f);
    }
    __syncthreads();

    if (tid < 2) {                                   // dense 352->2, relu
        float s = bo[tid];
        for (int i = 0; i < 352; ++i) s += x2[i] * wo[i * 2 + tid];
        out[g * 2 + tid] = fmaxf(s, 0.f);
    }
}

// =============================================================================
extern "C" void kernel_launch(void* const* d_in, const int* in_sizes, int n_in,
                              void* d_out, int out_size, void* d_ws, size_t ws_size,
                              hipStream_t stream) {
    const float* nf  = (const float*)d_in[0];
    const float* ef  = (const float*)d_in[1];
    const int*   ei  = (const int*)  d_in[2];
    const float* W0  = (const float*)d_in[3];  const float* b0 = (const float*)d_in[4];
    const float* W1  = (const float*)d_in[5];  const float* b1 = (const float*)d_in[6];
    const float* W2  = (const float*)d_in[7];  const float* b2 = (const float*)d_in[8];
    const float* W3  = (const float*)d_in[9];  const float* b3 = (const float*)d_in[10];
    const float* wc1 = (const float*)d_in[11]; const float* bc1 = (const float*)d_in[12];
    const float* wc2 = (const float*)d_in[13]; const float* bc2 = (const float*)d_in[14];
    const float* wo  = (const float*)d_in[15]; const float* bo  = (const float*)d_in[16];
    float* out = (float*)d_out;

    const int N = in_sizes[0] / 128;
    const int E = in_sizes[1] / 32;
    const int B = N / 1024;
    const int* src = ei;
    const int* dst = ei + E;

    float* ws     = (float*)d_ws;
    float* deg    = ws;                              // N
    float* e2n    = deg + N;                         // N*32
    float* pooled = e2n + (size_t)N * 32;            // N*160
    float* feat   = pooled + (size_t)N * 160;        // N*128 (97 used)

    const int TB = 256;
    auto blks = [](long n, int tb) { return (unsigned)((n + tb - 1) / tb); };

    init_deg_e2n<<<blks((long)N * 32, TB), TB, 0, stream>>>(deg, e2n, N);
    edge_e2n<<<blks((long)E * 32, TB), TB, 0, stream>>>(ef, dst, e2n, deg, E);

    // layer 0: 160 -> 32
    pooled_init0<<<blks((long)N * 160, TB), TB, 0, stream>>>(nf, e2n, pooled, N);
    scatter0<<<blks((long)E * 160, TB), TB, 0, stream>>>(nf, e2n, src, dst, pooled, E);
    gnn_gemm_kernel<160, 32><<<N / 128, 256, 0, stream>>>(pooled, W0, b0, deg, feat, 0);

    // layer 1: 32 -> 32
    pooled_initL<<<blks((long)N * 32, TB), TB, 0, stream>>>(feat, pooled, N, 0);
    scatterL<<<blks((long)E * 32, TB), TB, 0, stream>>>(feat, src, dst, pooled, E, 0);
    gnn_gemm_kernel<32, 32><<<N / 128, 256, 0, stream>>>(pooled, W1, b1, deg, feat, 32);

    // layer 2: 32 -> 32
    pooled_initL<<<blks((long)N * 32, TB), TB, 0, stream>>>(feat, pooled, N, 32);
    scatterL<<<blks((long)E * 32, TB), TB, 0, stream>>>(feat, src, dst, pooled, E, 32);
    gnn_gemm_kernel<32, 32><<<N / 128, 256, 0, stream>>>(pooled, W2, b2, deg, feat, 64);

    // layer 3: 32 -> 1
    pooled_initL<<<blks((long)N * 32, TB), TB, 0, stream>>>(feat, pooled, N, 64);
    scatterL<<<blks((long)E * 32, TB), TB, 0, stream>>>(feat, src, dst, pooled, E, 64);
    gnn_gemm_kernel<32, 1><<<N / 128, 256, 0, stream>>>(pooled, W3, b3, deg, feat, 96);

    // SortPool + conv tail
    tail_kernel<<<B, 128, 0, stream>>>(feat, wc1, bc1, wc2, bc2, wo, bo, out);
}
